// SelfAttention_19731079758426
// MI455X (gfx1250) — matrix operand order repaired and checked by reading, not compiled, not run on hardware
//
#include <hip/hip_runtime.h>

#define HID 2048
#define NH 16
#define HD 128
#define SQ 2048
#define BS 2

typedef __attribute__((ext_vector_type(16))) __bf16 v16bf;
typedef __attribute__((ext_vector_type(8))) float v8f;
typedef unsigned int u32x4 __attribute__((ext_vector_type(4)));
typedef int i32x8 __attribute__((ext_vector_type(8)));
typedef int i32x4 __attribute__((ext_vector_type(4)));

union Frag16 {
  v16bf v;
  uint4 q[2];
  unsigned short s[16];
};

__device__ __forceinline__ unsigned short f2bf(float f) {
  unsigned int u = __float_as_uint(f);
  u += 0x7FFFu + ((u >> 16) & 1u);
  return (unsigned short)(u >> 16);
}

__device__ __forceinline__ v8f zero8() {
  v8f z;
#pragma unroll
  for (int i = 0; i < 8; ++i) z[i] = 0.0f;
  return z;
}

__device__ __forceinline__ v8f wmma_bf16(v16bf a, v16bf b, v8f c) {
  return __builtin_amdgcn_wmma_f32_16x16x32_bf16(false, a, false, b, (short)0, c,
                                                 false, false);
}

// TDM: load a [128 rows x 32 cols] bf16 tile (row stride 2048 elems) into LDS.
// D# per CDNA5 ISA ch.8: group0 = {count/flags, lds_addr, gaddr_lo, gaddr_hi|type},
// group1 = dims/strides, groups 2-3 zero (2D tensor).
__device__ __forceinline__ void tdm_load_tile(const void* gptr, unsigned lds_off) {
  unsigned long long ga = (unsigned long long)gptr;
  u32x4 g0;
  g0.x = 1u;                                   // count=1, no gather/iterate
  g0.y = lds_off;                              // LDS byte address
  g0.z = (unsigned)ga;                         // global addr [31:0]
  g0.w = (unsigned)((ga >> 32) & 0x01FFFFFFu)  // global addr [56:32]
         | 0x80000000u;                        // type=2 (image)
  i32x8 g1;
  g1[0] = 0x00010000;          // data_size=1 (2 bytes/elem)
  g1[1] = (int)(2048u << 16);  // tensor_dim0[15:0]=2048 (bits 63:48)
  g1[2] = (int)(128u << 16);   // tensor_dim0[31:16]=0 | tensor_dim1[15:0]=128
  g1[3] = (int)(32u << 16);    // tensor_dim1[31:16]=0 | tile_dim0=32
  g1[4] = 128;                 // tile_dim1=128 | tile_dim2=0
  g1[5] = 2048;                // tensor_dim0_stride[31:0]
  g1[6] = 0;                   // stride hi | dim1_stride lo
  g1[7] = 0;                   // dim1_stride hi
  i32x4 z4 = {0, 0, 0, 0};
  i32x8 z8 = {0, 0, 0, 0, 0, 0, 0, 0};
  __builtin_amdgcn_tensor_load_to_lds(g0, g1, z4, z4, z8, 0);
}

// ---------------------------------------------------------------- convert x
__global__ void k_f32_to_bf16(const float* __restrict__ in,
                              unsigned short* __restrict__ out, int n4) {
  int i = blockIdx.x * blockDim.x + threadIdx.x;
  if (i >= n4) return;
  float4 v = reinterpret_cast<const float4*>(in)[i];
  uint2 p;
  p.x = (unsigned)f2bf(v.x) | ((unsigned)f2bf(v.y) << 16);
  p.y = (unsigned)f2bf(v.z) | ((unsigned)f2bf(v.w) << 16);
  reinterpret_cast<uint2*>(out)[i] = p;
}

// ------------------------------------------- transpose fp32[R][C] -> bf16[C][R]
__global__ void k_transpose_bf16(const float* __restrict__ in,
                                 unsigned short* __restrict__ out, int R, int C) {
  __shared__ float tile[32][33];
  int tx = threadIdx.x, ty = threadIdx.y;
  int x = blockIdx.x * 32 + tx;
#pragma unroll
  for (int j = 0; j < 4; ++j) {
    int y = blockIdx.y * 32 + ty + j * 8;
    tile[ty + j * 8][tx] = in[(size_t)y * C + x];
  }
  __syncthreads();
  int xo = blockIdx.y * 32 + tx;
#pragma unroll
  for (int j = 0; j < 4; ++j) {
    int yo = blockIdx.x * 32 + ty + j * 8;
    out[(size_t)yo * R + xo] = f2bf(tile[tx][ty + j * 8]);
  }
}

// -------------------------------------------------------------- QKV GEMM
// Block: 256 thr / 8 waves -> 128M x 128N tile. TDM double-buffers 128x32
// A and B tiles in LDS; each wave computes 32M x 64N (8 WMMAs / K-step).
__global__ void k_qkv_gemm(const unsigned short* __restrict__ A,
                           const unsigned short* __restrict__ Bw,
                           const float* __restrict__ bias,
                           unsigned short* __restrict__ Qb,
                           unsigned short* __restrict__ Kb,
                           unsigned short* __restrict__ Vtb) {
  __shared__ __align__(16) unsigned short smem[2][2][128 * 32];
  const int wid = threadIdx.x >> 5, lane = threadIdx.x & 31;
  const int half = lane >> 4, l16 = lane & 15;
  const int mloc = (wid & 3) * 32;
  const int nloc = (wid >> 2) * 64;
  const int mbase = blockIdx.x * 128;
  const int nbase = blockIdx.y * 128;
  v8f acc[2][4];
#pragma unroll
  for (int s = 0; s < 2; ++s)
#pragma unroll
    for (int t = 0; t < 4; ++t) acc[s][t] = zero8();

  const unsigned short* Ablk = A + (size_t)mbase * HID;
  const unsigned short* Bblk = Bw + (size_t)nbase * HID;

  if (wid == 0) {
    tdm_load_tile(Ablk, (unsigned)(unsigned long long)&smem[0][0][0]);
    tdm_load_tile(Bblk, (unsigned)(unsigned long long)&smem[0][1][0]);
    __builtin_amdgcn_s_wait_tensorcnt(0);
  }
  __syncthreads();

  int buf = 0;
  for (int kk = 0; kk < HID; kk += 32) {
    if (wid == 0 && kk + 32 < HID) {
      tdm_load_tile(Ablk + kk + 32,
                    (unsigned)(unsigned long long)&smem[buf ^ 1][0][0]);
      tdm_load_tile(Bblk + kk + 32,
                    (unsigned)(unsigned long long)&smem[buf ^ 1][1][0]);
    }
    const unsigned short* sA = &smem[buf][0][0];
    const unsigned short* sB = &smem[buf][1][0];
    Frag16 a0, a1;
    const unsigned short* pa0 = sA + (mloc + l16) * 32 + half * 8;
    const unsigned short* pa1 = sA + (mloc + 16 + l16) * 32 + half * 8;
    a0.q[0] = *reinterpret_cast<const uint4*>(pa0);
    a0.q[1] = *reinterpret_cast<const uint4*>(pa0 + 16);
    a1.q[0] = *reinterpret_cast<const uint4*>(pa1);
    a1.q[1] = *reinterpret_cast<const uint4*>(pa1 + 16);
#pragma unroll
    for (int t = 0; t < 4; ++t) {
      Frag16 bfr;
      const unsigned short* pb = sB + (nloc + t * 16 + l16) * 32 + half * 8;
      bfr.q[0] = *reinterpret_cast<const uint4*>(pb);
      bfr.q[1] = *reinterpret_cast<const uint4*>(pb + 16);
      acc[0][t] = wmma_bf16(a0.v, bfr.v, acc[0][t]);
      acc[1][t] = wmma_bf16(a1.v, bfr.v, acc[1][t]);
    }
    if (wid == 0) __builtin_amdgcn_s_wait_tensorcnt(0);
    __syncthreads();
    buf ^= 1;
  }

#pragma unroll
  for (int sub = 0; sub < 2; ++sub) {
#pragma unroll
    for (int t = 0; t < 4; ++t) {
      int n = nbase + nloc + t * 16 + l16;
      int sel = n >> 11;  // 0:q 1:k 2:v
      int rem = n & (HID - 1);
      int h = rem >> 7, d = rem & (HD - 1);
      float bv = bias[n];
#pragma unroll
      for (int r = 0; r < 8; ++r) {
        int m = mbase + mloc + sub * 16 + r + half * 8;
        int b = m >> 11, s = m & (SQ - 1);
        unsigned short o = f2bf(acc[sub][t][r] + bv);
        size_t bh = (size_t)(b * NH + h);
        if (sel == 0)
          Qb[(bh * SQ + s) * HD + d] = o;
        else if (sel == 1)
          Kb[(bh * SQ + s) * HD + d] = o;
        else
          Vtb[(bh * HD + d) * SQ + s] = o;  // V stored transposed per head
      }
    }
  }
}

// ------------------------------------------------------------ flash attention
__global__ void k_flash(const unsigned short* __restrict__ Qb,
                        const unsigned short* __restrict__ Kb,
                        const unsigned short* __restrict__ Vtb,
                        const int* __restrict__ amask,
                        unsigned short* __restrict__ Ob) {
  __shared__ __align__(16) unsigned short pbuf[4][512];
  const int wid = threadIdx.x >> 5, lane = threadIdx.x & 31;
  const int half = lane >> 4, l16 = lane & 15;
  const int bh = blockIdx.x, b = bh >> 4, h = bh & (NH - 1);
  const int qbase = (blockIdx.y * 4 + wid) * 16;
  const unsigned short* Qh = Qb + (size_t)bh * SQ * HD;
  const unsigned short* Kh = Kb + (size_t)bh * SQ * HD;
  const unsigned short* Vh = Vtb + (size_t)bh * HD * SQ;

  Frag16 qa[4];
  const unsigned short* qrow = Qh + (size_t)(qbase + l16) * HD;
#pragma unroll
  for (int kd = 0; kd < 4; ++kd) {
    int c0 = kd * 32 + half * 16;
    qa[kd].q[0] = *reinterpret_cast<const uint4*>(qrow + c0);
    qa[kd].q[1] = *reinterpret_cast<const uint4*>(qrow + c0 + 8);
  }

  v8f acc[8];
  float mrow[8], lrow[8];
#pragma unroll
  for (int dt = 0; dt < 8; ++dt) acc[dt] = zero8();
#pragma unroll
  for (int r = 0; r < 8; ++r) {
    mrow[r] = -1e30f;
    lrow[r] = 0.0f;
  }

  const float scale = 0.08838834764831845f;  // 1/sqrt(128)
  const int qend = qbase + 15;
  for (int kt = 0; kt <= qend; kt += 32) {
    v8f sf0 = zero8(), sf1 = zero8();
#pragma unroll
    for (int kd = 0; kd < 4; ++kd) {
      Frag16 kf0, kf1;
      const uint4* kp0 = reinterpret_cast<const uint4*>(
          Kh + (size_t)(kt + l16) * HD + kd * 32 + half * 16);
      const uint4* kp1 = reinterpret_cast<const uint4*>(
          Kh + (size_t)(kt + 16 + l16) * HD + kd * 32 + half * 16);
      kf0.q[0] = kp0[0]; kf0.q[1] = kp0[1];
      kf1.q[0] = kp1[0]; kf1.q[1] = kp1[1];
      sf0 = wmma_bf16(qa[kd].v, kf0.v, sf0);
      sf1 = wmma_bf16(qa[kd].v, kf1.v, sf1);
    }
    const int k0 = kt + l16, k1 = kt + 16 + l16;
    const int ok0 = amask[b * SQ + k0];
    const int ok1 = amask[b * SQ + k1];
    float tmax[8];
#pragma unroll
    for (int r = 0; r < 8; ++r) {
      int q = qbase + r + half * 8;
      float s0 = sf0[r] * scale;
      float s1 = sf1[r] * scale;
      if (k0 > q || !ok0) s0 = -1e30f;
      if (k1 > q || !ok1) s1 = -1e30f;
      sf0[r] = s0;
      sf1[r] = s1;
      float mx = fmaxf(s0, s1);
      mx = fmaxf(mx, __shfl_xor(mx, 1, 32));
      mx = fmaxf(mx, __shfl_xor(mx, 2, 32));
      mx = fmaxf(mx, __shfl_xor(mx, 4, 32));
      mx = fmaxf(mx, __shfl_xor(mx, 8, 32));
      tmax[r] = mx;
    }
    float alpha[8];
#pragma unroll
    for (int r = 0; r < 8; ++r) {
      float nm = fmaxf(mrow[r], tmax[r]);
      alpha[r] = __expf(mrow[r] - nm);
      mrow[r] = nm;
      float p0 = __expf(sf0[r] - nm);
      float p1 = __expf(sf1[r] - nm);
      sf0[r] = p0;
      sf1[r] = p1;
      float ps = p0 + p1;
      ps += __shfl_xor(ps, 1, 32);
      ps += __shfl_xor(ps, 2, 32);
      ps += __shfl_xor(ps, 4, 32);
      ps += __shfl_xor(ps, 8, 32);
      lrow[r] = lrow[r] * alpha[r] + ps;
    }
#pragma unroll
    for (int dt = 0; dt < 8; ++dt)
#pragma unroll
      for (int r = 0; r < 8; ++r) acc[dt][r] *= alpha[r];

    unsigned short* pw = pbuf[wid];
#pragma unroll
    for (int r = 0; r < 8; ++r) {
      int row = r + half * 8;
      pw[row * 32 + l16] = f2bf(sf0[r]);
      pw[row * 32 + 16 + l16] = f2bf(sf1[r]);
    }
    asm volatile("s_wait_dscnt 0x0" ::: "memory");
    Frag16 pf;
    {
      const unsigned short* prow = pw + l16 * 32 + half * 16;
      pf.q[0] = *reinterpret_cast<const uint4*>(prow);
      pf.q[1] = *reinterpret_cast<const uint4*>(prow + 8);
    }
#pragma unroll
    for (int dt = 0; dt < 8; ++dt) {
      Frag16 vf;
      const uint4* vp = reinterpret_cast<const uint4*>(
          Vh + (size_t)(dt * 16 + l16) * SQ + kt + half * 16);
      vf.q[0] = vp[0];
      vf.q[1] = vp[1];
      acc[dt] = wmma_bf16(pf.v, vf.v, acc[dt]);
    }
  }
#pragma unroll
  for (int r = 0; r < 8; ++r) lrow[r] = 1.0f / lrow[r];
#pragma unroll
  for (int dt = 0; dt < 8; ++dt) {
    int d = h * HD + dt * 16 + l16;
#pragma unroll
    for (int r = 0; r < 8; ++r) {
      int q = qbase + r + half * 8;
      Ob[((size_t)b * SQ + q) * HID + d] = f2bf(acc[dt][r] * lrow[r]);
    }
  }
}

// -------------------------------------------------------------- output GEMM
__global__ void k_out_gemm(const unsigned short* __restrict__ A,
                           const unsigned short* __restrict__ Bw,
                           const float* __restrict__ bias,
                           float* __restrict__ out) {
  __shared__ __align__(16) unsigned short smem[2][2][128 * 32];
  const int wid = threadIdx.x >> 5, lane = threadIdx.x & 31;
  const int half = lane >> 4, l16 = lane & 15;
  const int mloc = (wid & 3) * 32;
  const int nloc = (wid >> 2) * 64;
  const int mbase = blockIdx.x * 128;
  const int nbase = blockIdx.y * 128;
  v8f acc[2][4];
#pragma unroll
  for (int s = 0; s < 2; ++s)
#pragma unroll
    for (int t = 0; t < 4; ++t) acc[s][t] = zero8();

  const unsigned short* Ablk = A + (size_t)mbase * HID;
  const unsigned short* Bblk = Bw + (size_t)nbase * HID;

  if (wid == 0) {
    tdm_load_tile(Ablk, (unsigned)(unsigned long long)&smem[0][0][0]);
    tdm_load_tile(Bblk, (unsigned)(unsigned long long)&smem[0][1][0]);
    __builtin_amdgcn_s_wait_tensorcnt(0);
  }
  __syncthreads();

  int buf = 0;
  for (int kk = 0; kk < HID; kk += 32) {
    if (wid == 0 && kk + 32 < HID) {
      tdm_load_tile(Ablk + kk + 32,
                    (unsigned)(unsigned long long)&smem[buf ^ 1][0][0]);
      tdm_load_tile(Bblk + kk + 32,
                    (unsigned)(unsigned long long)&smem[buf ^ 1][1][0]);
    }
    const unsigned short* sA = &smem[buf][0][0];
    const unsigned short* sB = &smem[buf][1][0];
    Frag16 a0, a1;
    const unsigned short* pa0 = sA + (mloc + l16) * 32 + half * 8;
    const unsigned short* pa1 = sA + (mloc + 16 + l16) * 32 + half * 8;
    a0.q[0] = *reinterpret_cast<const uint4*>(pa0);
    a0.q[1] = *reinterpret_cast<const uint4*>(pa0 + 16);
    a1.q[0] = *reinterpret_cast<const uint4*>(pa1);
    a1.q[1] = *reinterpret_cast<const uint4*>(pa1 + 16);
#pragma unroll
    for (int t = 0; t < 4; ++t) {
      Frag16 bfr;
      const unsigned short* pb = sB + (nloc + t * 16 + l16) * 32 + half * 8;
      bfr.q[0] = *reinterpret_cast<const uint4*>(pb);
      bfr.q[1] = *reinterpret_cast<const uint4*>(pb + 16);
      acc[0][t] = wmma_bf16(a0.v, bfr.v, acc[0][t]);
      acc[1][t] = wmma_bf16(a1.v, bfr.v, acc[1][t]);
    }
    if (wid == 0) __builtin_amdgcn_s_wait_tensorcnt(0);
    __syncthreads();
    buf ^= 1;
  }

#pragma unroll
  for (int sub = 0; sub < 2; ++sub) {
#pragma unroll
    for (int t = 0; t < 4; ++t) {
      int n = nbase + nloc + t * 16 + l16;
      float bv = bias[n];
#pragma unroll
      for (int r = 0; r < 8; ++r) {
        int m = mbase + mloc + sub * 16 + r + half * 8;
        out[(size_t)m * HID + n] = acc[sub][t][r] + bv;
      }
    }
  }
}

extern "C" void kernel_launch(void* const* d_in, const int* in_sizes, int n_in,
                              void* d_out, int out_size, void* d_ws,
                              size_t ws_size, hipStream_t stream) {
  (void)in_sizes; (void)n_in; (void)out_size; (void)ws_size;
  const float* x = (const float*)d_in[0];
  const int* amask = (const int*)d_in[1];
  const float* Wqkv = (const float*)d_in[2];
  const float* bqkv = (const float*)d_in[3];
  const float* Wout = (const float*)d_in[4];
  const float* bout = (const float*)d_in[5];
  float* out = (float*)d_out;

  char* ws = (char*)d_ws;
  size_t o = 0;
  unsigned short* xb = (unsigned short*)(ws + o);    o += (size_t)BS * SQ * HID * 2;
  unsigned short* wqkvT = (unsigned short*)(ws + o); o += (size_t)3 * HID * HID * 2;
  unsigned short* woutT = (unsigned short*)(ws + o); o += (size_t)HID * HID * 2;
  unsigned short* Qb = (unsigned short*)(ws + o);    o += (size_t)BS * NH * SQ * HD * 2;
  unsigned short* Kb = (unsigned short*)(ws + o);    o += (size_t)BS * NH * SQ * HD * 2;
  unsigned short* Vtb = (unsigned short*)(ws + o);   o += (size_t)BS * NH * HD * SQ * 2;
  unsigned short* Ob = (unsigned short*)(ws + o);    o += (size_t)BS * SQ * HID * 2;

  const int n4 = BS * SQ * HID / 4;
  k_f32_to_bf16<<<(n4 + 255) / 256, 256, 0, stream>>>(x, xb, n4);
  k_transpose_bf16<<<dim3(3 * HID / 32, HID / 32), dim3(32, 8), 0, stream>>>(
      Wqkv, wqkvT, HID, 3 * HID);
  k_transpose_bf16<<<dim3(HID / 32, HID / 32), dim3(32, 8), 0, stream>>>(
      Wout, woutT, HID, HID);
  k_qkv_gemm<<<dim3(BS * SQ / 128, 3 * HID / 128), 256, 0, stream>>>(
      xb, wqkvT, bqkv, Qb, Kb, Vtb);
  k_flash<<<dim3(BS * NH, SQ / 64), 128, 0, stream>>>(Qb, Kb, Vtb, amask, Ob);
  k_out_gemm<<<dim3(BS * SQ / 128, HID / 128), 256, 0, stream>>>(Ob, woutT, bout,
                                                                 out);
}
